// CrossAttention_28183575396936
// MI455X (gfx1250) — matrix operand order repaired and checked by reading, not compiled, not run on hardware
//
#include <hip/hip_runtime.h>

typedef __bf16 bf16_t;
typedef bf16_t v16bf __attribute__((ext_vector_type(16)));
typedef float  v8f   __attribute__((ext_vector_type(8)));

union BfVec {
  v16bf bf;
  unsigned int u[8];
  uint4 q[2];
};

__device__ __forceinline__ unsigned short f2bf(float f) {
  unsigned int u = __builtin_bit_cast(unsigned int, f);
  u += 0x7FFFu + ((u >> 16) & 1u);          // round-to-nearest-even
  return (unsigned short)(u >> 16);
}

__device__ __forceinline__ unsigned int pack_bf(float a, float b) {
  return (unsigned int)f2bf(a) | ((unsigned int)f2bf(b) << 16);
}

__device__ __forceinline__ v8f vzero8() {
  v8f z = {0.f, 0.f, 0.f, 0.f, 0.f, 0.f, 0.f, 0.f};
  return z;
}

__device__ __forceinline__ v8f wmma_bf16(v16bf a, v16bf b, v8f c) {
  // (neg_a, A, neg_b, B, c_mod, C, reuse_a, reuse_b)
  return __builtin_amdgcn_wmma_f32_16x16x32_bf16(false, a, false, b, (short)0, c,
                                                 false, false);
}

// A-operand (16x32 bf16) fragment from row-major bf16 in LDS.
// A layout (matches B swizzle): lane l -> row l&15; lanes 0-15 hold K = k0..k0+15;
// lanes 16-31 hold K = k0+16..k0+31 (contiguous k order).
__device__ __forceinline__ v16bf load_a_lds(const unsigned short* base, int ld,
                                            int lane, int k0) {
  int row = lane & 15;
  int kb  = k0 + ((lane >> 4) << 4);
  const unsigned short* p = base + row * ld + kb;
  BfVec r;
  r.q[0] = *(const uint4*)(p);        // K = kb .. kb+7
  r.q[1] = *(const uint4*)(p + 8);    // K = kb+8 .. kb+15
  return r.bf;
}

// B-operand (32x16 bf16) fragment from a pre-swizzled tile of 256 dwords:
// entry[lane*8+v] packs (B[kt*32 + (lane>>4)*16 + 2v][n], B[..+2v+1][n]),
// n = nt*16 + (lane&15).  -> two coalesced b128 loads per lane.
__device__ __forceinline__ v16bf load_b_swz(const unsigned int* tile, int lane) {
  BfVec r;
  const uint4* p = (const uint4*)(tile + lane * 8);
  r.q[0] = p[0];
  r.q[1] = p[1];
  return r.bf;
}

// ---------------------------------------------------------------------------
// fp32 [K][N] weight -> bf16 B-operand-swizzled tiles.
// ---------------------------------------------------------------------------
__global__ void conv_wt(const float* __restrict__ src, unsigned int* __restrict__ dst,
                        int K, int N) {
  int Nt = N >> 4;
  int total = (K >> 5) * Nt * 256;
  for (int idx = blockIdx.x * blockDim.x + threadIdx.x; idx < total;
       idx += gridDim.x * blockDim.x) {
    int v    = idx & 7;
    int lane = (idx >> 3) & 31;
    int tile = idx >> 8;
    int nt = tile % Nt;
    int kt = tile / Nt;
    int k = kt * 32 + ((lane >> 4) << 4) + 2 * v;
    int n = nt * 16 + (lane & 15);
    dst[idx] = pack_bf(src[(size_t)k * N + n], src[(size_t)(k + 1) * N + n]);
  }
}

// ---------------------------------------------------------------------------
// K,V projections: context[b,0:77,0:768] @ Wk/Wv -> bf16, scattered into
// B-operand swizzled form:
//   Kswz[b][h]: matrix Kt[d][j]  (64 x 80),  kt in 0..1, ntj in 0..4
//   Vswz[b][h]: matrix V [j][d]  (96 x 64),  kt in 0..2, ntd in 0..3
// Buffers are pre-zeroed, so padded rows/cols (j >= 77) stay zero.
// Grid: 80 = B*5 row-tiles; 8 waves, wave w handles head w's 64 columns.
// ---------------------------------------------------------------------------
__global__ __launch_bounds__(256, 1) void kv_proj(
    const float* __restrict__ ctx, const unsigned int* __restrict__ WkS,
    const unsigned int* __restrict__ WvS, unsigned int* __restrict__ Kswz,
    unsigned int* __restrict__ Vswz) {
  int b = blockIdx.x / 5, jt = blockIdx.x % 5;
  int tid = threadIdx.x, lane = tid & 31, w = tid >> 5;
  __shared__ __align__(16) unsigned short lds_c[16 * 768];

  for (int i = tid; i < 16 * 768; i += 256) {
    int r = i / 768, c = i % 768;
    int j = jt * 16 + r;
    if (j > 76) j = 76;  // clamp; garbage rows are masked on store
    lds_c[i] = f2bf(ctx[((size_t)b * 77 + j) * 768 + c]);
  }
  __syncthreads();

  int hgrp = lane >> 4;
  for (int nt4 = 0; nt4 < 4; ++nt4) {
    int nt = w * 4 + nt4;  // column tile in [0,32): columns of head w
    v8f ka = vzero8(), va = vzero8();
    for (int kt = 0; kt < 24; ++kt) {
      v16bf a  = load_a_lds(lds_c, 768, lane, kt * 32);
      v16bf bk = load_b_swz(WkS + (size_t)(kt * 32 + nt) * 256, lane);
      v16bf bv = load_b_swz(WvS + (size_t)(kt * 32 + nt) * 256, lane);
      ka = wmma_bf16(a, bk, ka);
      va = wmma_bf16(a, bv, va);
    }
    int c0 = nt * 16 + (lane & 15);  // column in [0,512)
    int d  = c0 & 63;                // within-head dim

    // ---- K: value = Kt[d][j]; pair (d, d+1) lives in neighbor lane ----
#pragma unroll
    for (int r = 0; r < 8; ++r) {
      float partner = __shfl_xor(ka[r], 1, 32);
      int j = jt * 16 + hgrp * 8 + r;
      if (((c0 & 1) == 0) && j < 77) {
        unsigned int pk = pack_bf(ka[r], partner);
        int kt2 = d >> 5, rem = d & 31;
        int lane2 = ((rem >> 4) << 4) + (j & 15);
        int v = (rem & 15) >> 1;
        size_t idx =
            ((((size_t)b * 8 + w) * 2 + kt2) * 5 + (j >> 4)) * 256 + lane2 * 8 + v;
        Kswz[idx] = pk;
      }
    }
    // ---- V: value = V[j][d]; pair (j, j+1) = consecutive accum comps ----
#pragma unroll
    for (int s = 0; s < 4; ++s) {
      int j = jt * 16 + hgrp * 8 + 2 * s;  // even
      if (j < 77) {
        float v1 = (j + 1 < 77) ? va[2 * s + 1] : 0.f;
        unsigned int pk = pack_bf(va[2 * s], v1);
        int kt2 = j >> 5, rem = j & 31;
        int lane2 = ((rem >> 4) << 4) + (d & 15);
        int v = (rem & 15) >> 1;
        size_t idx =
            ((((size_t)b * 8 + w) * 3 + kt2) * 4 + (d >> 4)) * 256 + lane2 * 8 + v;
        Vswz[idx] = pk;
      }
    }
  }
}

// ---------------------------------------------------------------------------
// Fused: Q-proj + QK^T + softmax + attn*V + out-proj for one (b, 16-row) block.
// 8 waves = 8 heads. LDS is the C/D->A layout converter.
// ---------------------------------------------------------------------------
__global__ __launch_bounds__(256, 1) void attn_fused(
    const float* __restrict__ x, const unsigned int* __restrict__ WqS,
    const unsigned int* __restrict__ WoS, const unsigned int* __restrict__ Kswz,
    const unsigned int* __restrict__ Vswz, const float* __restrict__ bo,
    float* __restrict__ out) {
  const int wg = blockIdx.x;
  const int b = wg >> 8;      // 4096/16 = 256 row tiles per batch
  const int r0 = (wg & 255) * 16;
  const int tid = threadIdx.x;
  const int lane = tid & 31;
  const int w = tid >> 5;     // wave id == head id
  const int hgrp = lane >> 4;
  const int cl = lane & 15;

  __shared__ __align__(16) unsigned short lds_x[16 * 320];     // x block, bf16
  __shared__ __align__(16) unsigned short lds_q[16 * 512];     // Q, later O
  __shared__ __align__(16) unsigned short lds_attn[8 * 16 * 96];

  // ---- stage x[b, r0:r0+16, :] (contiguous) as bf16 ----
  const float* xblk = x + ((size_t)b * 4096 + r0) * 320;
  for (int i = tid; i < 16 * 320; i += 256) lds_x[i] = f2bf(xblk[i]);
  __syncthreads();

  // ---- Q projection: wave w computes columns [w*64, w*64+64) ----
  for (int nt4 = 0; nt4 < 4; ++nt4) {
    int nt = w * 4 + nt4;  // in [0,32)
    v8f acc = vzero8();
    for (int kt = 0; kt < 10; ++kt) {
      v16bf a  = load_a_lds(lds_x, 320, lane, kt * 32);
      v16bf bm = load_b_swz(WqS + (size_t)(kt * 32 + nt) * 256, lane);
      acc = wmma_bf16(a, bm, acc);
    }
    int col = nt * 16 + cl;
#pragma unroll
    for (int r = 0; r < 8; ++r) lds_q[(r + hgrp * 8) * 512 + col] = f2bf(acc[r]);
  }

  // ---- sim = Q K^T * scale (wave-local: reads only its own Q columns) ----
  float sims[5][8];
  const unsigned int* Kb = Kswz + ((size_t)b * 8 + w) * 10 * 256;
  for (int nt = 0; nt < 5; ++nt) {
    v8f acc = vzero8();
    for (int kt = 0; kt < 2; ++kt) {
      v16bf a  = load_a_lds(lds_q, 512, lane, w * 64 + kt * 32);
      v16bf bm = load_b_swz(Kb + (kt * 5 + nt) * 256, lane);
      acc = wmma_bf16(a, bm, acc);
    }
#pragma unroll
    for (int r = 0; r < 8; ++r) sims[nt][r] = acc[r] * 0.125f;  // 64^-0.5
  }
  // mask j >= 77
#pragma unroll
  for (int nt = 0; nt < 5; ++nt) {
    if (nt * 16 + cl >= 77) {
#pragma unroll
      for (int r = 0; r < 8; ++r) sims[nt][r] = -1e30f;
    }
  }

  // ---- softmax per row (row data spread over 16 lanes of a half-wave) ----
  float attnp[5][8];
#pragma unroll
  for (int r = 0; r < 8; ++r) {
    float m = sims[0][r];
#pragma unroll
    for (int nt = 1; nt < 5; ++nt) m = fmaxf(m, sims[nt][r]);
#pragma unroll
    for (int s = 1; s < 16; s <<= 1) m = fmaxf(m, __shfl_xor(m, s, 32));
    float e[5], sum = 0.f;
#pragma unroll
    for (int nt = 0; nt < 5; ++nt) { e[nt] = __expf(sims[nt][r] - m); sum += e[nt]; }
#pragma unroll
    for (int s = 1; s < 16; s <<= 1) sum += __shfl_xor(sum, s, 32);
    float inv = 1.0f / sum;
#pragma unroll
    for (int nt = 0; nt < 5; ++nt) attnp[nt][r] = e[nt] * inv;
  }

  // ---- attn -> LDS (row-major 16x96 bf16, per head; pad 80..95 = 0) ----
  unsigned short* at = lds_attn + w * (16 * 96);
  for (int nt = 0; nt < 5; ++nt) {
    int col = nt * 16 + cl;
#pragma unroll
    for (int r = 0; r < 8; ++r) at[(r + hgrp * 8) * 96 + col] = f2bf(attnp[nt][r]);
  }
  {
    unsigned int* atz = (unsigned int*)at;  // 48 dwords per row
    for (int i = lane; i < 16 * 8; i += 32) atz[(i >> 3) * 48 + 40 + (i & 7)] = 0;
  }

  // ---- O = attn @ V (16x64 per head) ----
  v8f oacc[4] = {vzero8(), vzero8(), vzero8(), vzero8()};
  const unsigned int* Vb = Vswz + ((size_t)b * 8 + w) * 12 * 256;
  for (int kt = 0; kt < 3; ++kt) {
    v16bf a = load_a_lds(at, 96, lane, kt * 32);
    for (int nt = 0; nt < 4; ++nt) {
      v16bf bm = load_b_swz(Vb + (kt * 4 + nt) * 256, lane);
      oacc[nt] = wmma_bf16(a, bm, oacc[nt]);
    }
  }
  // reuse lds_q as the concatenated-heads O buffer (same columns this wave owned)
  for (int nt = 0; nt < 4; ++nt) {
    int col = w * 64 + nt * 16 + cl;
#pragma unroll
    for (int r = 0; r < 8; ++r) lds_q[(r + hgrp * 8) * 512 + col] = f2bf(oacc[nt][r]);
  }
  __syncthreads();

  // ---- out = O @ Wo + bo, N tiles split round-robin over waves ----
  float* outb = out + ((size_t)b * 4096 + r0) * 320;
  for (int nt = w; nt < 20; nt += 8) {
    v8f acc = vzero8();
    for (int kt = 0; kt < 16; ++kt) {
      v16bf a  = load_a_lds(lds_q, 512, lane, kt * 32);
      v16bf bm = load_b_swz(WoS + (size_t)(kt * 20 + nt) * 256, lane);
      acc = wmma_bf16(a, bm, acc);
    }
    int col = nt * 16 + cl;
    float bias = bo[col];
#pragma unroll
    for (int r = 0; r < 8; ++r) outb[(r + hgrp * 8) * 320 + col] = acc[r] + bias;
  }
}

extern "C" void kernel_launch(void* const* d_in, const int* in_sizes, int n_in,
                              void* d_out, int out_size, void* d_ws, size_t ws_size,
                              hipStream_t stream) {
  (void)in_sizes; (void)n_in; (void)out_size; (void)ws_size;
  const float* x   = (const float*)d_in[0];
  const float* ctx = (const float*)d_in[1];
  const float* Wq  = (const float*)d_in[2];
  const float* Wk  = (const float*)d_in[3];
  const float* Wv  = (const float*)d_in[4];
  const float* Wo  = (const float*)d_in[5];
  const float* bo  = (const float*)d_in[6];
  float* out = (float*)d_out;

  // workspace layout (dwords): ~5.1 MB total
  unsigned int* ws   = (unsigned int*)d_ws;
  unsigned int* WqS  = ws;                  // 320*512/2  = 81920
  unsigned int* WkS  = WqS + 81920;         // 768*512/2  = 196608
  unsigned int* WvS  = WkS + 196608;        // 196608
  unsigned int* WoS  = WvS + 196608;        // 512*320/2  = 81920
  unsigned int* Kswz = WoS + 81920;         // 16*8*2*5*256 = 327680
  unsigned int* Vswz = Kswz + 327680;       // 16*8*3*4*256 = 393216

  conv_wt<<<320, 256, 0, stream>>>(Wq, WqS, 320, 512);
  conv_wt<<<768, 256, 0, stream>>>(Wk, WkS, 768, 512);
  conv_wt<<<768, 256, 0, stream>>>(Wv, WvS, 768, 512);
  conv_wt<<<320, 256, 0, stream>>>(Wo, WoS, 512, 320);
  hipMemsetAsync(Kswz, 0, (size_t)(327680 + 393216) * sizeof(unsigned int), stream);
  kv_proj<<<80, 256, 0, stream>>>(ctx, WkS, WvS, Kswz, Vswz);
  attn_fused<<<16 * 256, 256, 0, stream>>>(x, WqS, WoS, Kswz, Vswz, bo, out);
}
